// NetsVocab_76012331204829
// MI455X (gfx1250) — compile-verified, matrix operand order checked by reading, not run on hardware
//
#include <hip/hip_runtime.h>

typedef __bf16 bf16_t;
typedef bf16_t v16bf __attribute__((ext_vector_type(16)));
typedef bf16_t v8bf  __attribute__((ext_vector_type(8)));
typedef bf16_t v2bf  __attribute__((ext_vector_type(2)));
typedef float  v8f   __attribute__((ext_vector_type(8)));
typedef float  v4f   __attribute__((ext_vector_type(4)));

constexpr int kNB   = 8192;   // batch (nBBox)
constexpr int kF    = 4096;   // feature dim
constexpr int kH1   = 64;
constexpr int kH2   = 32;
constexpr int kNW   = 16;     // active words
constexpr int kTILE = 128;    // batch rows per block (8 waves x 16)
constexpr int kCH   = 64;     // K-chunk (two WMMA K-steps per staging round)

// Load a 32x16 (k x n) bf16 B fragment from LDS ([n][k] layout, row stride = kCH or kH1).
__device__ __forceinline__ v16bf load_bfrag(const bf16_t* p)
{
    union { v16bf v; v8bf h[2]; } u;
    u.h[0] = *(const v8bf*)(p);        // K run {0..7}   (per-lane half)
    u.h[1] = *(const v8bf*)(p + 16);   // K run {16..23} (per-lane half)
    return u.v;
}

// Stage a [kCH x 64] f32 chunk of W1 into LDS transposed [n][kk] as bf16,
// using packed cvt + ds_store_b32 (each thread: 2 k-rows x 8 n).
__device__ __forceinline__ void stage_w1(const float* __restrict__ W1w,
                                         bf16_t* __restrict__ dst,
                                         int k0, int tid)
{
    const int kk2 = (tid >> 3) * 2;          // even k within chunk: 0..62
    const int n0  = (tid & 7) * 8;           // 8 consecutive n
    const float* s0 = W1w + (size_t)(k0 + kk2) * kH1 + n0;
    v4f r0a = *(const v4f*)(s0);
    v4f r0b = *(const v4f*)(s0 + 4);
    v4f r1a = *(const v4f*)(s0 + kH1);
    v4f r1b = *(const v4f*)(s0 + kH1 + 4);
    #pragma unroll
    for (int j = 0; j < 4; ++j) {
        v2bf p0 = { (bf16_t)r0a[j], (bf16_t)r1a[j] };
        *(v2bf*)&dst[(n0 + j) * kCH + kk2] = p0;
        v2bf p1 = { (bf16_t)r0b[j], (bf16_t)r1b[j] };
        *(v2bf*)&dst[(n0 + 4 + j) * kCH + kk2] = p1;
    }
}

// ---------------------------------------------------------------------------
// Kernel 1: for word w = blockIdx.y, batch tile = blockIdx.x, compute
//   sig[w][row] = sigmoid( relu(relu(x W1 + b1) W2 + b2) . W3 + b3 )
// ---------------------------------------------------------------------------
__global__ __launch_bounds__(256)
void mlp_word_kernel(const float* __restrict__ x,
                     const int*   __restrict__ words,
                     const float* __restrict__ W1,
                     const float* __restrict__ b1,
                     const float* __restrict__ W2,
                     const float* __restrict__ b2,
                     const float* __restrict__ W3,
                     const float* __restrict__ b3,
                     float*       __restrict__ sig)   // [kNW][kNB]
{
    __shared__ bf16_t ldsB1[2][kH1 * kCH];    // double-buffered W1 chunk [n][kk]
    __shared__ bf16_t ldsW2[kH2 * kH1];       // W2 transposed [n][k]
    __shared__ bf16_t ldsH1[8 * 16 * kH1];    // per-wave h1 tile [M][K]

    const int tid  = threadIdx.x;
    const int lane = tid & 31;
    const int wav  = tid >> 5;
    const int c16  = lane & 15;
    const int hlf  = lane >> 4;

    const int wIdx    = blockIdx.y;
    const int word    = words[wIdx];
    const int rowBase = blockIdx.x * kTILE + wav * 16;

    const float* W1w = W1 + (size_t)word * kF * kH1;
    const float* b1w = b1 + word * kH1;
    const float* W2w = W2 + word * kH1 * kH2;
    const float* b2w = b2 + word * kH2;
    const float* W3w = W3 + word * kH2;
    const float  b3w = b3[word];

    // ---- stage W2 (64x32 f32, k-major) -> ldsW2[n*64 + k] bf16 (one-time) --
    {
        const int f0 = tid * 8;               // 2048 floats / 256 threads
        const int k  = f0 / kH2;
        const int n0 = f0 % kH2;              // 8 consecutive n, same k
        v4f a = *(const v4f*)(W2w + f0);
        v4f b = *(const v4f*)(W2w + f0 + 4);
        #pragma unroll
        for (int j = 0; j < 4; ++j) ldsW2[(n0 + j) * kH1 + k]     = (bf16_t)a[j];
        #pragma unroll
        for (int j = 0; j < 4; ++j) ldsW2[(n0 + 4 + j) * kH1 + k] = (bf16_t)b[j];
    }

    // ---- layer 1: x[16x4096] (wave tile) @ W1[4096x64] -> acc[4] -----------
    v8f acc[4] = {};
    const float* xrow = x + (size_t)(rowBase + c16) * kF;

    stage_w1(W1w, ldsB1[0], 0, tid);          // prologue: chunk 0 -> buffer 0

    for (int c = 0; c < kF / kCH; ++c) {
        const int k0 = c * kCH;
        __syncthreads();                      // staging of chunk c visible
        const bf16_t* cur = ldsB1[c & 1];
        if (k0 + kCH < kF)                    // overlap: stage chunk c+1
            stage_w1(W1w, ldsB1[(c & 1) ^ 1], k0 + kCH, tid);

        // A fragments for both 32-K subchunks (f32 -> bf16 in-register)
        const float* ap = xrow + k0 + hlf * 8;
        v4f a0 = *(const v4f*)(ap);      v4f a1 = *(const v4f*)(ap + 4);
        v4f a2 = *(const v4f*)(ap + 16); v4f a3 = *(const v4f*)(ap + 20);
        v4f a4 = *(const v4f*)(ap + 32); v4f a5 = *(const v4f*)(ap + 36);
        v4f a6 = *(const v4f*)(ap + 48); v4f a7 = *(const v4f*)(ap + 52);
        if (k0 + 2 * kCH < kF) __builtin_prefetch(ap + 2 * kCH, 0, 0);
        v16bf af0, af1;
        #pragma unroll
        for (int j = 0; j < 4; ++j) {
            af0[j]      = (bf16_t)a0[j]; af0[4 + j]  = (bf16_t)a1[j];
            af0[8 + j]  = (bf16_t)a2[j]; af0[12 + j] = (bf16_t)a3[j];
            af1[j]      = (bf16_t)a4[j]; af1[4 + j]  = (bf16_t)a5[j];
            af1[8 + j]  = (bf16_t)a6[j]; af1[12 + j] = (bf16_t)a7[j];
        }

        // subchunk 0: load all 4 B fragments, then burst 4 WMMAs
        v16bf bfr[4];
        #pragma unroll
        for (int nt = 0; nt < 4; ++nt)
            bfr[nt] = load_bfrag(&cur[(nt * 16 + c16) * kCH + hlf * 8]);
        #pragma unroll
        for (int nt = 0; nt < 4; ++nt)
            acc[nt] = __builtin_amdgcn_wmma_f32_16x16x32_bf16(
                false, af0, false, bfr[nt], (short)0, acc[nt], false, false);

        // subchunk 1
        #pragma unroll
        for (int nt = 0; nt < 4; ++nt)
            bfr[nt] = load_bfrag(&cur[(nt * 16 + c16) * kCH + 32 + hlf * 8]);
        #pragma unroll
        for (int nt = 0; nt < 4; ++nt)
            acc[nt] = __builtin_amdgcn_wmma_f32_16x16x32_bf16(
                false, af1, false, bfr[nt], (short)0, acc[nt], false, false);
    }

    // ---- bias + ReLU, store h1 tile to LDS as bf16 [M][K] (per-wave) -------
    bf16_t* h1base = &ldsH1[wav * 16 * kH1];
    #pragma unroll
    for (int nt = 0; nt < 4; ++nt) {
        const int n = nt * 16 + c16;          // C layout: N = lane&15
        const float bias = b1w[n];
        #pragma unroll
        for (int r = 0; r < 8; ++r) {         // M = r + 8*hlf
            float v = acc[nt][r] + bias;
            v = v > 0.0f ? v : 0.0f;
            h1base[(r + hlf * 8) * kH1 + n] = (bf16_t)v;
        }
    }
    __syncthreads();

    // ---- layer 2: h1[16x64] @ W2[64x32] -> d[2] (4 WMMAs, hoisted loads) ---
    v16bf a2f[2], b2f[2][2];
    #pragma unroll
    for (int kc = 0; kc < 2; ++kc) {
        a2f[kc] = load_bfrag(&h1base[c16 * kH1 + kc * 32 + hlf * 8]);
        #pragma unroll
        for (int nt = 0; nt < 2; ++nt)
            b2f[kc][nt] = load_bfrag(&ldsW2[(nt * 16 + c16) * kH1 + kc * 32 + hlf * 8]);
    }
    v8f d[2] = {};
    #pragma unroll
    for (int kc = 0; kc < 2; ++kc)
        #pragma unroll
        for (int nt = 0; nt < 2; ++nt)
            d[nt] = __builtin_amdgcn_wmma_f32_16x16x32_bf16(
                false, a2f[kc], false, b2f[kc][nt], (short)0, d[nt], false, false);

    // ---- layer 3 (dot with W3) + bias2/ReLU + sigmoid ----------------------
    const float w3a = W3w[c16];          // for d[0]: N = c16
    const float w3b = W3w[16 + c16];     // for d[1]: N = 16 + c16
    const float b2a = b2w[c16];
    const float b2b = b2w[16 + c16];
    #pragma unroll
    for (int r = 0; r < 8; ++r) {
        float h2a = d[0][r] + b2a; h2a = h2a > 0.0f ? h2a : 0.0f;
        float h2b = d[1][r] + b2b; h2b = h2b > 0.0f ? h2b : 0.0f;
        float p = h2a * w3a + h2b * w3b;
        // reduce across the 16 lanes of each half (wave32 butterfly)
        #pragma unroll
        for (int off = 1; off < 16; off <<= 1)
            p += __shfl_xor(p, off, 32);
        if (c16 == 0) {
            const int row = rowBase + r + hlf * 8;   // M = r + 8*hlf
            const float logit = p + b3w;
            sig[wIdx * kNB + row] = 1.0f / (1.0f + __expf(-logit));
        }
    }
}

// ---------------------------------------------------------------------------
// Kernel 2: out[b] = prod_w sig[w][b]
// ---------------------------------------------------------------------------
__global__ __launch_bounds__(256)
void prod_kernel(const float* __restrict__ sig, float* __restrict__ out)
{
    const int b = blockIdx.x * blockDim.x + threadIdx.x;
    if (b >= kNB) return;
    float p = 1.0f;
    #pragma unroll
    for (int w = 0; w < kNW; ++w) p *= sig[w * kNB + b];
    out[b] = p;
}

extern "C" void kernel_launch(void* const* d_in, const int* in_sizes, int n_in,
                              void* d_out, int out_size, void* d_ws, size_t ws_size,
                              hipStream_t stream)
{
    // setup_inputs order: nBBox, x, words, W1, b1, W2, b2, W3, b3
    const float* x     = (const float*)d_in[1];
    const int*   words = (const int*)  d_in[2];
    const float* W1    = (const float*)d_in[3];
    const float* b1    = (const float*)d_in[4];
    const float* W2    = (const float*)d_in[5];
    const float* b2    = (const float*)d_in[6];
    const float* W3    = (const float*)d_in[7];
    const float* b3    = (const float*)d_in[8];
    float* out = (float*)d_out;
    float* sig = (float*)d_ws;      // kNW * kNB floats = 512 KB scratch

    dim3 grid(kNB / kTILE, kNW);    // 64 batch tiles x 16 words
    mlp_word_kernel<<<grid, 256, 0, stream>>>(x, words, W1, b1, W2, b2, W3, b3, sig);
    prod_kernel<<<kNB / 256, 256, 0, stream>>>(sig, out);
}
